// Net_49830210568754
// MI455X (gfx1250) — matrix operand order repaired and checked by reading, not compiled, not run on hardware
//
#include <hip/hip_runtime.h>
#include <hip/hip_bf16.h>

// ---------------- CDNA5 WMMA plumbing (gfx1250, wave32) ----------------
typedef __attribute__((ext_vector_type(16))) __bf16 v16bf;
typedef __attribute__((ext_vector_type(8)))  float  v8f;

union TileU { v16bf v; uint4 q[2]; };

// A-tile: 16x32 bf16 from row-major [16 x ld] at (row0,k0)=(0,0).
// ISA layout: lanes 0-15 -> M=lane, K=0..7 & 16..23 ; lanes 16-31 -> M=lane-16, K=8..15 & 24..31
__device__ __forceinline__ v16bf load_tileA(const __hip_bfloat16* base, int ld, int lane) {
  int row = lane & 15;
  int k0  = (lane & 16) ? 8 : 0;
  const __hip_bfloat16* r = base + (size_t)row * ld + k0;
  TileU u;
  u.q[0] = *reinterpret_cast<const uint4*>(r);
  u.q[1] = *reinterpret_cast<const uint4*>(r + 16);
  return u.v;
}

// B-tile: K=32 x N=16 taken from row-major weight W[N][K] at (n0,k0); ld = K stride.
// lane n (0-15): col n, K=0..7 & 16..23 ; lane n+16: col n, K=8..15 & 24..31
__device__ __forceinline__ v16bf load_tileB(const __hip_bfloat16* w, int ldw, int lane) {
  int n  = lane & 15;
  int kk = (lane & 16) ? 8 : 0;
  const __hip_bfloat16* r = w + (size_t)n * ldw + kk;
  TileU u;
  u.q[0] = *reinterpret_cast<const uint4*>(r);
  u.q[1] = *reinterpret_cast<const uint4*>(r + 16);
  return u.v;
}

// C/D f32 16x16: lane -> column (lane&15), VGPR v -> row v + ((lane&16)?8:0)
__device__ __forceinline__ v8f load_tileC(const float* in, int ld, int lane) {
  int col = lane & 15, rb = (lane & 16) ? 8 : 0;
  v8f c;
#pragma unroll
  for (int v = 0; v < 8; ++v) c[v] = in[(size_t)(rb + v) * ld + col];
  return c;
}
__device__ __forceinline__ void store_tileD(float* out, int ld, int lane, v8f d) {
  int col = lane & 15, rb = (lane & 16) ? 8 : 0;
#pragma unroll
  for (int v = 0; v < 8; ++v) out[(size_t)(rb + v) * ld + col] = d[v];
}

__device__ __forceinline__ v8f wmma_bf16(v16bf a, v16bf b, v8f c) {
  return __builtin_amdgcn_wmma_f32_16x16x32_bf16(false, a, false, b, (short)0, c, false, false);
}

__device__ __forceinline__ float sigmoidf_(float x) { return 1.f / (1.f + __expf(-x)); }
__device__ __forceinline__ float sgf_(float x)      { return 0.5f * tanhf(0.5f * x) + 0.5f; }

// ---------------- problem dims ----------------
constexpr int Hn = 256, Bn = 16, Tn = 400, Ln = 50, NCn = 2000, NCPn = 2016, NCAn = 64;

// ---------------- conversion kernels ----------------
__global__ void cvt_bf16_kernel(const float* __restrict__ s, __hip_bfloat16* __restrict__ d, int n) {
  int i = blockIdx.x * blockDim.x + threadIdx.x;
  if (i < n) d[i] = __float2bfloat16(s[i]);
}
__global__ void cvt_pad_rows_kernel(const float* __restrict__ s, __hip_bfloat16* __restrict__ d,
                                    int rows, int cs, int cd) {
  int i = blockIdx.x * blockDim.x + threadIdx.x;
  if (i >= rows * cd) return;
  int r = i / cd, c = i - r * cd;
  d[i] = __float2bfloat16(c < cs ? s[(size_t)r * cs + c] : 0.f);
}
// src [B][T][C] f32 -> dst [T][B][Cp] bf16 (zero-pad C..Cp)
__global__ void cvt_transpose_pad_kernel(const float* __restrict__ s, __hip_bfloat16* __restrict__ d,
                                         int B_, int T_, int C, int Cp) {
  int i = blockIdx.x * blockDim.x + threadIdx.x;
  if (i >= T_ * B_ * Cp) return;
  int t = i / (B_ * Cp); int rem = i - t * B_ * Cp; int b = rem / Cp; int c = rem - b * Cp;
  d[i] = __float2bfloat16(c < C ? s[((size_t)b * T_ + t) * C + c] : 0.f);
}

// ---------------- generic row-tile GEMM: O[row16-tile] = A @ W^T + b1 + b2 ----------------
__global__ void gemm_rows_kernel(const __hip_bfloat16* __restrict__ A, size_t a_rstride, int lda,
                                 const __hip_bfloat16* __restrict__ W, int ldw,
                                 const float* __restrict__ b1, const float* __restrict__ b2,
                                 float* __restrict__ O, size_t o_rstride, int ldo,
                                 int K, int tilesN) {
  int lane = threadIdx.x & 31, wv = threadIdx.x >> 5;
  int tile = blockIdx.x * (blockDim.x >> 5) + wv;
  if (tile >= tilesN) return;           // wave-uniform
  int n0 = tile * 16;
  const __hip_bfloat16* At = A + blockIdx.y * a_rstride;
  float* Ot = O + blockIdx.y * o_rstride;
  float bb = (b1 ? b1[n0 + (lane & 15)] : 0.f) + (b2 ? b2[n0 + (lane & 15)] : 0.f);
  v8f acc;
#pragma unroll
  for (int v = 0; v < 8; ++v) acc[v] = bb;
  for (int k = 0; k < K; k += 32)
    acc = wmma_bf16(load_tileA(At + k, lda, lane), load_tileB(W + (size_t)n0 * ldw + k, ldw, lane), acc);
  store_tileD(Ot + n0, ldo, lane, acc);
}

// ---------------- LSTM input projection: preact[dir][t][b][4H] = x[t] @ Wih^T + bih + bhh ----------------
__global__ void inproj_kernel(const __hip_bfloat16* __restrict__ x, int K,
                              const __hip_bfloat16* __restrict__ wih,
                              const float* __restrict__ bih, const float* __restrict__ bhh,
                              float* __restrict__ preact, int T_) {
  int lane = threadIdx.x & 31, wv = threadIdx.x >> 5;
  int t = blockIdx.y, dir = blockIdx.z;
  int n0 = (blockIdx.x * 4 + wv) * 16;
  const __hip_bfloat16* A  = x + (size_t)t * 16 * K;
  const __hip_bfloat16* Wd = wih + (size_t)dir * 1024 * K + (size_t)n0 * K;
  float bb = bih[dir * 1024 + n0 + (lane & 15)] + bhh[dir * 1024 + n0 + (lane & 15)];
  v8f acc;
#pragma unroll
  for (int v = 0; v < 8; ++v) acc[v] = bb;
  for (int k = 0; k < K; k += 32)
    acc = wmma_bf16(load_tileA(A + k, K, lane), load_tileB(Wd + k, K, lane), acc);
  store_tileD(preact + ((size_t)dir * T_ + t) * 16 * 1024 + n0, 1024, lane, acc);
}

// ---------------- LSTM recurrence: one persistent 32-wave block per direction ----------------
__global__ void __launch_bounds__(1024) lstm_recur_kernel(
    const float* __restrict__ preact, const __hip_bfloat16* __restrict__ whh,
    const int* __restrict__ length, __hip_bfloat16* __restrict__ hout, int T_) {
  extern __shared__ char smem[];
  __hip_bfloat16* h_bf = (__hip_bfloat16*)smem;            // 16x256 bf16 (8KB)
  float* h_f   = (float*)(smem + 8192);                    // 16KB
  float* c_f   = h_f + 4096;                               // 16KB
  float* gates = c_f + 4096;                               // 16x1024 f32 (64KB)
  int*   lenS  = (int*)(gates + 16384);

  int dir = blockIdx.x;
  const float* pre = preact + (size_t)dir * T_ * 16 * 1024;
  const __hip_bfloat16* W = whh + (size_t)dir * 1024 * 256;
  int tid = threadIdx.x, lane = tid & 31, wv = tid >> 5;

  for (int i = tid; i < 4096; i += 1024) { h_f[i] = 0.f; c_f[i] = 0.f; h_bf[i] = __float2bfloat16(0.f); }
  if (tid < 16) lenS[tid] = length[tid];
  __syncthreads();

  for (int step = 0; step < T_; ++step) {
    int t = dir ? (T_ - 1 - step) : step;
    const float* pt = pre + (size_t)t * 16 * 1024;
    for (int i2 = 0; i2 < 2; ++i2) {
      int n0 = (wv * 2 + i2) * 16;
      const __hip_bfloat16* wr = W + (size_t)n0 * 256;
      v8f acc = load_tileC(pt + n0, 1024, lane);
      for (int k = 0; k < 256; k += 32) {
        if (k + 32 < 256) __builtin_prefetch(wr + (size_t)(lane & 15) * 256 + k + 32, 0, 1);
        acc = wmma_bf16(load_tileA(h_bf + k, 256, lane), load_tileB(wr + k, 256, lane), acc);
      }
      store_tileD(gates + n0, 1024, lane, acc);
    }
    __syncthreads();
    for (int i = tid; i < 4096; i += 1024) {
      int b = i >> 8, j = i & 255;
      float gi = gates[b * 1024 + j],        gf = gates[b * 1024 + 256 + j];
      float gg = gates[b * 1024 + 512 + j],  go = gates[b * 1024 + 768 + j];
      float c2 = sigmoidf_(gf) * c_f[i] + sigmoidf_(gi) * tanhf(gg);
      float h2 = sigmoidf_(go) * tanhf(c2);
      bool m = t < lenS[b];
      float hn = m ? h2 : h_f[i];
      c_f[i] = m ? c2 : c_f[i];
      h_f[i] = hn;
      h_bf[i] = __float2bfloat16(hn);
      hout[((size_t)t * 16 + b) * 512 + dir * 256 + j] = __float2bfloat16(m ? h2 : 0.f);
    }
    __syncthreads();
  }
}

// ---------------- Decoder: one persistent 1024-thread block, all state in LDS ----------------
__global__ void __launch_bounds__(1024) decoder_kernel(
    const __hip_bfloat16* __restrict__ hpack, const float* __restrict__ Hh,
    const float* __restrict__ danda, const __hip_bfloat16* __restrict__ gtb,
    const int* __restrict__ length, const float* __restrict__ attw,
    const __hip_bfloat16* watts,
    const __hip_bfloat16* wsy, const __hip_bfloat16* wgy, const float* bgy,
    const __hip_bfloat16* wyy, const float* byy,
    const __hip_bfloat16* wss, const __hip_bfloat16* wgs, const float* bgs,
    const __hip_bfloat16* wys, const float* bys,
    const __hip_bfloat16* wsn, const __hip_bfloat16* wnn, const float* bnn,
    const __hip_bfloat16* wwn, const float* bwn,
    const __hip_bfloat16* wnda, const float* bnda,
    float* __restrict__ ys_out, float* __restrict__ Ys_out) {
  extern __shared__ char smem[];
  float* s_f    = (float*)smem;           // [16][256]
  float* c_f    = s_f    + 4096;
  float* n_f    = c_f    + 4096;
  float* y1_f   = n_f    + 4096;
  float* n1_f   = y1_f   + 4096;
  float* g_f    = n1_f   + 4096;          // [16][512]
  float* sW     = g_f    + 8192;          // [16][512]
  float* gates  = sW     + 8192;          // [16][1024]
  float* ealpha = gates  + 16384;         // [16][400]
  float* red    = ealpha + 6400;          // [16][64]
  float* red2   = red    + 1024;          // [16][64]
  __hip_bfloat16* s_bf  = (__hip_bfloat16*)(red2 + 1024);  // [16][256]
  __hip_bfloat16* n_bf  = s_bf  + 4096;
  __hip_bfloat16* y1_bf = n_bf  + 4096;
  __hip_bfloat16* n1_bf = y1_bf + 4096;                    // [16][256]
  __hip_bfloat16* g_bf  = n1_bf + 4096;                    // [16][512]
  float* attwS = (float*)(g_bf + 8192);                    // [512]
  int*   lenS  = (int*)(attwS + 512);

  int tid = threadIdx.x, lane = tid & 31, wv = tid >> 5;
  for (int i = tid; i < 4096; i += 1024) {
    s_f[i] = 0.f; c_f[i] = 0.f; n_f[i] = 0.f;
    s_bf[i] = __float2bfloat16(0.f); n_bf[i] = __float2bfloat16(0.f);
  }
  for (int i = tid; i < 512; i += 1024) attwS[i] = attw[i];
  if (tid < 16) lenS[tid] = length[tid];
  __syncthreads();

  for (int l = 0; l < Ln; ++l) {
    // ---- phase 1: sW = s @ w_atts^T  (32 tiles, one per wave) ----
    {
      int n0 = wv * 16;
      v8f acc;
#pragma unroll
      for (int v = 0; v < 8; ++v) acc[v] = 0.f;
      for (int k = 0; k < 256; k += 32)
        acc = wmma_bf16(load_tileA(s_bf + k, 256, lane),
                        load_tileB(watts + (size_t)n0 * 256 + k, 256, lane), acc);
      store_tileD(sW + n0, 512, lane, acc);
    }
    __syncthreads();
    // ---- phase 2: e[b][t] = dot(tanh(sW[b]+Hh[t][b]), attw) for valid t ----
    for (int idx = tid; idx < 16 * Tn; idx += 1024) {
      int b = idx / Tn, t = idx - b * Tn;
      if (t < lenS[b]) {
        const float* hh = Hh + (size_t)t * 8192 + b * 512;
        const float* sw = sW + b * 512;
        float e = 0.f;
        for (int j = 0; j < 512; ++j) e += tanhf(sw[j] + hh[j]) * attwS[j];
        ealpha[b * Tn + t] = e;
      }
    }
    __syncthreads();
    // ---- phase 3: masked softmax over t (64 threads per batch row) ----
    {
      int b = tid >> 6, sub = tid & 63;
      int len = lenS[b];
      float mx = -3.0e38f;
      for (int t = sub; t < len; t += 64) mx = fmaxf(mx, ealpha[b * Tn + t]);
      red[b * 64 + sub] = mx;
      __syncthreads();
      if (sub == 0) { float m = -3.0e38f; for (int i = 0; i < 64; ++i) m = fmaxf(m, red[b * 64 + i]); red2[b * 64] = m; }
      __syncthreads();
      float M = red2[b * 64];
      float sm = 0.f;
      for (int t = sub; t < len; t += 64) sm += __expf(ealpha[b * Tn + t] - M);
      red[b * 64 + sub] = sm;
      __syncthreads();
      if (sub == 0) { float s = 0.f; for (int i = 0; i < 64; ++i) s += red[b * 64 + i]; red2[b * 64 + 1] = s; }
      __syncthreads();
      float inv = 1.f / red2[b * 64 + 1];
      for (int t = sub; t < Tn; t += 64)
        ealpha[b * Tn + t] = (t < len) ? __expf(ealpha[b * Tn + t] - M) * inv : 0.f;
    }
    __syncthreads();
    // ---- phase 4: g[b][j] = sum_t alpha * hpack[t][b][j] ----
    for (int idx = tid; idx < 8192; idx += 1024) {
      int b = idx >> 9, j = idx & 511;
      const float* al = ealpha + b * Tn;
      const __hip_bfloat16* hp = hpack + b * 512 + j;
      float acc = 0.f;
      for (int t = 0; t < Tn; ++t) acc += al[t] * __bfloat162float(hp[(size_t)t * 8192]);
      g_f[idx] = acc;
      g_bf[idx] = __float2bfloat16(acc);
    }
    __syncthreads();
    // ---- phase 5: y1 = s@w_sy^T+g@w_gy^T+b_gy (waves 0-15); n1 = s@w_sn^T+n@w_nn^T+b_nn+danda (16-31) ----
    if (wv < 16) {
      int n0 = wv * 16;
      float bb = bgy[n0 + (lane & 15)];
      v8f acc;
#pragma unroll
      for (int v = 0; v < 8; ++v) acc[v] = bb;
      for (int k = 0; k < 256; k += 32)
        acc = wmma_bf16(load_tileA(s_bf + k, 256, lane), load_tileB(wsy + (size_t)n0 * 256 + k, 256, lane), acc);
      for (int k = 0; k < 512; k += 32)
        acc = wmma_bf16(load_tileA(g_bf + k, 512, lane), load_tileB(wgy + (size_t)n0 * 512 + k, 512, lane), acc);
      store_tileD(y1_f + n0, 256, lane, acc);
    } else {
      int n0 = (wv - 16) * 16;
      v8f acc = load_tileC(danda + n0, 256, lane);
      float bb = bnn[n0 + (lane & 15)];
#pragma unroll
      for (int v = 0; v < 8; ++v) acc[v] += bb;
      for (int k = 0; k < 256; k += 32)
        acc = wmma_bf16(load_tileA(s_bf + k, 256, lane), load_tileB(wsn + (size_t)n0 * 256 + k, 256, lane), acc);
      for (int k = 0; k < 256; k += 32)
        acc = wmma_bf16(load_tileA(n_bf + k, 256, lane), load_tileB(wnn + (size_t)n0 * 256 + k, 256, lane), acc);
      store_tileD(n1_f + n0, 256, lane, acc);
    }
    __syncthreads();
    for (int i = tid; i < 4096; i += 1024) {
      float a = tanhf(y1_f[i]); y1_f[i] = a; y1_bf[i] = __float2bfloat16(a);
      float b2 = tanhf(n1_f[i]); n1_f[i] = b2; n1_bf[i] = __float2bfloat16(b2);
    }
    __syncthreads();
    // ---- phase 6a: Yd = n_old @ w_nda^T + b_nda -> Ys_out (waves 0-3) ----
    if (wv < 4) {
      int n0 = wv * 16;
      float bb = bnda[n0 + (lane & 15)];
      v8f acc;
#pragma unroll
      for (int v = 0; v < 8; ++v) acc[v] = bb;
      for (int k = 0; k < 256; k += 32)
        acc = wmma_bf16(load_tileA(n_bf + k, 256, lane), load_tileB(wnda + (size_t)n0 * 256 + k, 256, lane), acc);
      int col = lane & 15, rb = (lane & 16) ? 8 : 0;
#pragma unroll
      for (int v = 0; v < 8; ++v)
        Ys_out[(size_t)(rb + v) * (Ln * NCAn) + l * NCAn + n0 + col] = acc[v];
    }
    // ---- phase 6b: gates = s@w_ss^T + g@w_gs^T + gt@w_ys^T + b_gs + b_ys (64 tiles) ----
    {
      const __hip_bfloat16* gt_l = gtb + (size_t)l * 16 * NCPn;
      for (int i2 = 0; i2 < 2; ++i2) {
        int n0 = (wv * 2 + i2) * 16;
        float bb = bgs[n0 + (lane & 15)] + bys[n0 + (lane & 15)];
        v8f acc;
#pragma unroll
        for (int v = 0; v < 8; ++v) acc[v] = bb;
        for (int k = 0; k < 256; k += 32)
          acc = wmma_bf16(load_tileA(s_bf + k, 256, lane), load_tileB(wss + (size_t)n0 * 256 + k, 256, lane), acc);
        for (int k = 0; k < 512; k += 32)
          acc = wmma_bf16(load_tileA(g_bf + k, 512, lane), load_tileB(wgs + (size_t)n0 * 512 + k, 512, lane), acc);
        for (int k = 0; k < NCPn; k += 32)
          acc = wmma_bf16(load_tileA(gt_l + k, NCPn, lane), load_tileB(wys + (size_t)n0 * NCPn + k, NCPn, lane), acc);
        store_tileD(gates + n0, 1024, lane, acc);
      }
    }
    __syncthreads();
    // ---- phase 7: ys = y1 @ w_yy^T + b_yy (125 tiles); n2 = n1 @ w_wn^T + b_wn (waves 0-15) ----
    for (int tile = wv; tile < 125; tile += 32) {
      int n0 = tile * 16;
      float bb = byy[n0 + (lane & 15)];
      v8f acc;
#pragma unroll
      for (int v = 0; v < 8; ++v) acc[v] = bb;
      for (int k = 0; k < 256; k += 32)
        acc = wmma_bf16(load_tileA(y1_bf + k, 256, lane), load_tileB(wyy + (size_t)n0 * 256 + k, 256, lane), acc);
      int col = lane & 15, rb = (lane & 16) ? 8 : 0;
#pragma unroll
      for (int v = 0; v < 8; ++v)
        ys_out[(size_t)(rb + v) * (Ln * NCn) + l * NCn + n0 + col] = acc[v];
    }
    if (wv < 16) {
      int n0 = wv * 16;
      float bb = bwn[n0 + (lane & 15)];
      v8f acc;
#pragma unroll
      for (int v = 0; v < 8; ++v) acc[v] = bb;
      for (int k = 0; k < 256; k += 32)
        acc = wmma_bf16(load_tileA(n1_bf + k, 256, lane), load_tileB(wwn + (size_t)n0 * 256 + k, 256, lane), acc);
      store_tileD(n_f + n0, 256, lane, acc);
    }
    __syncthreads();
    // ---- phase 8: state update (sg-gated LSTM cell) ----
    for (int i = tid; i < 4096; i += 1024) {
      int b = i >> 8, j = i & 255;
      float gi = gates[b * 1024 + j],       gf = gates[b * 1024 + 256 + j];
      float gg = gates[b * 1024 + 512 + j], go = gates[b * 1024 + 768 + j];
      float c2 = sgf_(gf) * c_f[i] + sgf_(gi) * tanhf(gg);
      float s2 = sgf_(go) * tanhf(c2);
      c_f[i] = c2; s_f[i] = s2;
      s_bf[i] = __float2bfloat16(s2);
      n_bf[i] = __float2bfloat16(n_f[i]);
    }
    __syncthreads();
  }
}

// ---------------- host orchestration ----------------
extern "C" void kernel_launch(void* const* d_in, const int* in_sizes, int n_in,
                              void* d_out, int out_size, void* d_ws, size_t ws_size,
                              hipStream_t stream) {
  (void)in_sizes; (void)n_in; (void)out_size; (void)ws_size;
  const float* data   = (const float*)d_in[0];
  const int*   length = (const int*)d_in[1];
  const float* gt     = (const float*)d_in[2];
  const float* da     = (const float*)d_in[3];
  const float* w_ih0  = (const float*)d_in[4];
  const float* w_ih12 = (const float*)d_in[5];
  const float* w_hh   = (const float*)d_in[6];
  const float* b_ih   = (const float*)d_in[7];
  const float* b_hh   = (const float*)d_in[8];
  const float* w_attw = (const float*)d_in[9];
  const float* w_atts = (const float*)d_in[10];
  const float* w_atth = (const float*)d_in[11];
  const float* b_atth = (const float*)d_in[12];
  const float* w_sy   = (const float*)d_in[13];
  const float* w_gy   = (const float*)d_in[14];
  const float* b_gy   = (const float*)d_in[15];
  const float* w_yy   = (const float*)d_in[16];
  const float* b_yy   = (const float*)d_in[17];
  const float* w_ss   = (const float*)d_in[18];
  const float* w_gs   = (const float*)d_in[19];
  const float* b_gs   = (const float*)d_in[20];
  const float* w_ys   = (const float*)d_in[21];
  const float* b_ys   = (const float*)d_in[22];
  const float* w_sn   = (const float*)d_in[23];
  const float* w_nn   = (const float*)d_in[24];
  const float* b_nn   = (const float*)d_in[25];
  const float* w_wn   = (const float*)d_in[26];
  const float* b_wn   = (const float*)d_in[27];
  const float* w_dan  = (const float*)d_in[28];
  const float* b_dan  = (const float*)d_in[29];
  const float* w_nda  = (const float*)d_in[30];
  const float* b_nda  = (const float*)d_in[31];

  char* ws = (char*)d_ws;
  size_t off = 0;
  auto alloc = [&](size_t bytes) { char* p = ws + off; off = (off + bytes + 255) & ~(size_t)255; return p; };
  auto bfp   = [&](size_t e) { return (__hip_bfloat16*)alloc(e * 2); };
  auto f32p  = [&](size_t e) { return (float*)alloc(e * 4); };

  __hip_bfloat16* x0     = bfp((size_t)Tn * 16 * 128);          // padded layer-0 input [T][16][128]
  __hip_bfloat16* wih0b  = bfp((size_t)2 * 1024 * 128);
  __hip_bfloat16* wih12b = bfp((size_t)2 * 2 * 1024 * 512);
  __hip_bfloat16* whhb   = bfp((size_t)3 * 2 * 1024 * 256);
  __hip_bfloat16* wattsb = bfp((size_t)512 * 256);
  __hip_bfloat16* watthb = bfp((size_t)512 * 512);
  __hip_bfloat16* wsyb   = bfp((size_t)256 * 256);
  __hip_bfloat16* wgyb   = bfp((size_t)256 * 512);
  __hip_bfloat16* wyyb   = bfp((size_t)NCn * 256);
  __hip_bfloat16* wssb   = bfp((size_t)1024 * 256);
  __hip_bfloat16* wgsb   = bfp((size_t)1024 * 512);
  __hip_bfloat16* wysb   = bfp((size_t)1024 * NCPn);
  __hip_bfloat16* wsnb   = bfp((size_t)256 * 256);
  __hip_bfloat16* wnnb   = bfp((size_t)256 * 256);
  __hip_bfloat16* wwnb   = bfp((size_t)256 * 256);
  __hip_bfloat16* wdanb  = bfp((size_t)256 * 64);
  __hip_bfloat16* wndab  = bfp((size_t)64 * 256);
  __hip_bfloat16* gtb    = bfp((size_t)Ln * 16 * NCPn);
  __hip_bfloat16* dab    = bfp((size_t)16 * 64);
  float* preact = f32p((size_t)2 * Tn * 16 * 1024);
  __hip_bfloat16* hbufA = bfp((size_t)Tn * 16 * 512);
  __hip_bfloat16* hbufB = bfp((size_t)Tn * 16 * 512);
  float* Hhb   = f32p((size_t)Tn * 16 * 512);
  float* danda = f32p((size_t)16 * 256);

  constexpr unsigned RECUR_LDS = 8192 + 16384 + 16384 + 65536 + 64;
  constexpr unsigned DEC_LDS   = (4096 * 5 + 8192 * 2 + 16384 + 6400 + 1024 * 2) * 4 /*f32*/
                               + (4096 * 4 + 8192) * 2 /*bf16*/ + 512 * 4 + 64;
  (void)hipFuncSetAttribute(reinterpret_cast<const void*>(&lstm_recur_kernel),
                            hipFuncAttributeMaxDynamicSharedMemorySize, RECUR_LDS);
  (void)hipFuncSetAttribute(reinterpret_cast<const void*>(&decoder_kernel),
                            hipFuncAttributeMaxDynamicSharedMemorySize, DEC_LDS);

  auto cvt = [&](const float* s, __hip_bfloat16* d, int n) {
    cvt_bf16_kernel<<<dim3((n + 255) / 256), 256, 0, stream>>>(s, d, n);
  };
  // weight / input conversion to bf16 (K padded to 32-multiples where needed)
  cvt_transpose_pad_kernel<<<dim3((Tn * 16 * 128 + 255) / 256), 256, 0, stream>>>(data, x0, 16, Tn, 120, 128);
  cvt_pad_rows_kernel<<<dim3((2048 * 128 + 255) / 256), 256, 0, stream>>>(w_ih0, wih0b, 2048, 120, 128);
  cvt(w_ih12, wih12b, 2 * 2 * 1024 * 512);
  cvt(w_hh,   whhb,   3 * 2 * 1024 * 256);
  cvt(w_atts, wattsb, 512 * 256);
  cvt(w_atth, watthb, 512 * 512);
  cvt(w_sy,   wsyb,   256 * 256);
  cvt(w_gy,   wgyb,   256 * 512);
  cvt(w_yy,   wyyb,   NCn * 256);
  cvt(w_ss,   wssb,   1024 * 256);
  cvt(w_gs,   wgsb,   1024 * 512);
  cvt_pad_rows_kernel<<<dim3((1024 * NCPn + 255) / 256), 256, 0, stream>>>(w_ys, wysb, 1024, NCn, NCPn);
  cvt(w_sn,   wsnb,   256 * 256);
  cvt(w_nn,   wnnb,   256 * 256);
  cvt(w_wn,   wwnb,   256 * 256);
  cvt(w_dan,  wdanb,  256 * 64);
  cvt(w_nda,  wndab,  64 * 256);
  cvt_transpose_pad_kernel<<<dim3((Ln * 16 * NCPn + 255) / 256), 256, 0, stream>>>(gt, gtb, 16, Ln, NCn, NCPn);
  cvt(da, dab, 16 * 64);

  // 3-layer bidirectional LSTM: parallel input projection + persistent recurrence
  for (int l = 0; l < 3; ++l) {
    const __hip_bfloat16* x; int K;
    if (l == 0) { x = x0; K = 128; }
    else        { x = (l == 1) ? hbufA : hbufB; K = 512; }
    const __hip_bfloat16* wih = (l == 0) ? wih0b : (wih12b + (size_t)(l - 1) * 2 * 1024 * 512);
    __hip_bfloat16* hout = (l == 1) ? hbufB : hbufA;
    inproj_kernel<<<dim3(16, Tn, 2), 128, 0, stream>>>(x, K, wih, b_ih + l * 2048, b_hh + l * 2048, preact, Tn);
    lstm_recur_kernel<<<dim3(2), 1024, RECUR_LDS, stream>>>(preact, whhb + (size_t)l * 2 * 1024 * 256,
                                                            length, hout, Tn);
  }
  // hpack = hbufA (layer 2 output). Loop-invariant projections:
  gemm_rows_kernel<<<dim3(8, Tn), 128, 0, stream>>>(hbufA, (size_t)16 * 512, 512, watthb, 512,
                                                    b_atth, nullptr, Hhb, (size_t)16 * 512, 512, 512, 32);
  gemm_rows_kernel<<<dim3(4, 1), 128, 0, stream>>>(dab, 0, 64, wdanb, 64,
                                                   b_dan, nullptr, danda, 0, 256, 64, 16);

  float* ys_out = (float*)d_out;
  float* Ys_out = ys_out + (size_t)16 * Ln * NCn;
  decoder_kernel<<<dim3(1), 1024, DEC_LDS, stream>>>(
      hbufA, Hhb, danda, gtb, length, w_attw, wattsb,
      wsyb, wgyb, b_gy, wyyb, b_yy, wssb, wgsb, b_gs, wysb, b_ys,
      wsnb, wnnb, b_nn, wwnb, b_wn, wndab, b_nda, ys_out, Ys_out);
}
